// Encoder_21071109554474
// MI455X (gfx1250) — matrix operand order repaired and checked by reading, not compiled, not run on hardware
//
#include <hip/hip_runtime.h>

typedef __attribute__((ext_vector_type(16))) _Float16 v16h;
typedef __attribute__((ext_vector_type(8)))  _Float16 v8h;
typedef __attribute__((ext_vector_type(8)))  float    v8f;

#define NF 26
#define HID1 24
#define HID2 8
#define SEQLEN 5
#define ROWSTRIDE (SEQLEN * NF)   // 130 floats per batch row of x

__device__ __forceinline__ float fsig(float x) {
  // sigmoid(x) = 1 / (1 + e^-x), native v_exp_f32 + v_rcp_f32
  return __builtin_amdgcn_rcpf(1.0f + __expf(-x));
}
__device__ __forceinline__ float ftanh(float x) {
  float e = __expf(-2.0f * x);
  return (1.0f - e) * __builtin_amdgcn_rcpf(1.0f + e);
}
__device__ __forceinline__ v8f bcast8(float v) {
  v8f r;
#pragma unroll
  for (int i = 0; i < 8; ++i) r[i] = v;
  return r;
}
__device__ __forceinline__ v16h zero16h() {
  v16h r;
#pragma unroll
  for (int i = 0; i < 16; ++i) r[i] = (_Float16)0.0f;
  return r;
}

// D = A(16x32 f16) * B(32x16 f16) + C(16x16 f32)
#define WMMA_F16(A, B, C) \
  __builtin_amdgcn_wmma_f32_16x16x32_f16(false, (A), false, (B), (short)0, (C), false, false)

__global__ __launch_bounds__(256) void lstm2_wmma_kernel(
    const float* __restrict__ x,
    const float* __restrict__ w_ih1, const float* __restrict__ w_hh1,
    const float* __restrict__ b_ih1, const float* __restrict__ b_hh1,
    const float* __restrict__ w_ih2, const float* __restrict__ w_hh2,
    const float* __restrict__ b_ih2, const float* __restrict__ b_hh2,
    float* __restrict__ out, int batch)
{
  // Weights in f16, gate-padded layout [N][K], K contiguous (B-fragment friendly).
  // Layer1: gates padded 24 -> 32 cols each (N = 128, 8 tiles of 16).
  // Layer2: gates padded  8 -> 16 cols each (N =  64, 4 tiles of 16).
  __shared__ _Float16 sWih1[128][32];
  __shared__ _Float16 sWhh1[128][32];
  __shared__ _Float16 sWih2[64][32];
  __shared__ _Float16 sWhh2[64][32];
  __shared__ float    sB1[128];
  __shared__ float    sB2[64];
  // Per-wave A-matrix staging for the recurrent h feedback [M=16][K=32] f16.
  __shared__ _Float16 sH1[8][16][32];
  __shared__ _Float16 sH2[8][16][32];

  const int tid = threadIdx.x;

  // ---------------- stage weights into LDS (zero pads first) ----------------
  for (int i = tid; i < 128 * 32; i += 256) {
    (&sWih1[0][0])[i] = (_Float16)0.0f;
    (&sWhh1[0][0])[i] = (_Float16)0.0f;
  }
  for (int i = tid; i < 64 * 32; i += 256) {
    (&sWih2[0][0])[i] = (_Float16)0.0f;
    (&sWhh2[0][0])[i] = (_Float16)0.0f;
  }
  if (tid < 128) sB1[tid] = 0.0f;
  if (tid < 64)  sB2[tid] = 0.0f;
  __syncthreads();

  for (int i = tid; i < 96 * NF; i += 256) {          // w_ih1 [96][26]
    int r = i / NF, k = i % NF;
    sWih1[(r / HID1) * 32 + (r % HID1)][k] = (_Float16)w_ih1[i];
  }
  for (int i = tid; i < 96 * HID1; i += 256) {        // w_hh1 [96][24]
    int r = i / HID1, k = i % HID1;
    sWhh1[(r / HID1) * 32 + (r % HID1)][k] = (_Float16)w_hh1[i];
  }
  for (int i = tid; i < 32 * HID1; i += 256) {        // w_ih2 [32][24]
    int r = i / HID1, k = i % HID1;
    sWih2[(r / HID2) * 16 + (r % HID2)][k] = (_Float16)w_ih2[i];
  }
  for (int i = tid; i < 32 * HID2; i += 256) {        // w_hh2 [32][8]
    int r = i / HID2, k = i % HID2;
    sWhh2[(r / HID2) * 16 + (r % HID2)][k] = (_Float16)w_hh2[i];
  }
  if (tid < 96) sB1[(tid / HID1) * 32 + (tid % HID1)] = b_ih1[tid] + b_hh1[tid];
  if (tid < 32) sB2[(tid / HID2) * 16 + (tid % HID2)] = b_ih2[tid] + b_hh2[tid];
  __syncthreads();

  // ---------------- per-wave setup (wave32: 16 batch rows / wave) -----------
  const int wave = tid >> 5;
  const int lane = tid & 31;
  const int ln   = lane & 15;   // N column within tile / M row for A loads
  const int lh   = lane >> 4;   // lane half selects K half (B) / K chunks (A)
  const int b0   = blockIdx.x * 128 + wave * 16;

  _Float16 (*h1st)[32] = sH1[wave];
  _Float16 (*h2st)[32] = sH2[wave];

  // zero staging (K pads must be exactly 0; also gives h(0)=0)
  for (int i = lane; i < 16 * 32; i += 32) {
    (&h1st[0][0])[i] = (_Float16)0.0f;
    (&h2st[0][0])[i] = (_Float16)0.0f;
  }
  __builtin_amdgcn_wave_barrier();

  // B fragment: lane ln = column N, lanes 0-15 hold K0-7,K16-23; 16-31 hold K8-15,K24-31
  auto loadB = [&](const _Float16 (*W)[32], int tile) -> v16h {
    const _Float16* row = &W[tile * 16 + ln][0];
    v8h lo = *(const v8h*)(row + lh * 8);
    v8h hi = *(const v8h*)(row + 16 + lh * 8);
    v16h r;
#pragma unroll
    for (int i = 0; i < 8; ++i) { r[i] = lo[i]; r[8 + i] = hi[i]; }
    return r;
  };
  // A fragment from row-major staging [16][32]:
  // lane<16: K0-7 then K16-23 ; lane>=16: K8-15 then K24-31 (row M = ln)
  auto loadA = [&](const _Float16 (*S)[32]) -> v16h {
    const _Float16* row = &S[ln][0];
    v8h lo = *(const v8h*)(row + lh * 8);
    v8h hi = *(const v8h*)(row + 16 + lh * 8);
    v16h r;
#pragma unroll
    for (int i = 0; i < 8; ++i) { r[i] = lo[i]; r[8 + i] = hi[i]; }
    return r;
  };

  // Only the recurrent-critical weights stay resident in VGPRs (96 regs);
  // the input-side weights (fWih1/fWih2) are re-read from LDS at use sites
  // to keep total allocation under 256 VGPRs (avoids s_set_vgpr_msb churn).
  v16h fWhh1[8], fWhh2[4];
#pragma unroll
  for (int p = 0; p < 8; ++p) fWhh1[p] = loadB(sWhh1, p);
#pragma unroll
  for (int p = 0; p < 4; ++p) fWhh2[p] = loadB(sWhh2, p);

  float b1s[8], b2s[4];
#pragma unroll
  for (int p = 0; p < 8; ++p) b1s[p] = sB1[p * 16 + ln];
#pragma unroll
  for (int p = 0; p < 4; ++p) b2s[p] = sB2[p * 16 + ln];

  const int xrow = (b0 + ln < batch) ? (b0 + ln) : (batch - 1);
  const float* xbase = x + (size_t)xrow * ROWSTRIDE;

  v8f  c1t0 = bcast8(0.0f), c1t1 = bcast8(0.0f), c2t = bcast8(0.0f);
  v16h h1A = zero16h(), h2A = zero16h();
  float h2last[8];
#pragma unroll
  for (int e = 0; e < 8; ++e) h2last[e] = 0.0f;

#pragma unroll
  for (int t = 0; t < SEQLEN; ++t) {
    // ---- x A-fragment straight from global, clamped addresses (no EXEC
    // divergence). Pad lanes read finite garbage; the zero pad rows of the
    // B matrices annihilate it in the WMMA.
    const float* xp = xbase + t * NF;
    v16h xAt;
#pragma unroll
    for (int i = 0; i < 8; ++i) {
      xAt[i] = (_Float16)xp[lh * 8 + i];            // K 0..7 / 8..15, always valid
      int k2 = 16 + lh * 8 + i;                     // K 16..23 / 24..31
      xAt[8 + i] = (_Float16)xp[(k2 < NF) ? k2 : (NF - 1)];
    }

    // ---- layer 1: two column halves (hidden j=0..15 and j=16..23) ----
#pragma unroll
    for (int hh = 0; hh < 2; ++hh) {
      v8f gi = bcast8(b1s[0 + hh]);
      gi = WMMA_F16(xAt, loadB(sWih1, 0 + hh), gi);
      gi = WMMA_F16(h1A, fWhh1[0 + hh], gi);
      v8f gf = bcast8(b1s[2 + hh]);
      gf = WMMA_F16(xAt, loadB(sWih1, 2 + hh), gf);
      gf = WMMA_F16(h1A, fWhh1[2 + hh], gf);
      v8f gg = bcast8(b1s[4 + hh]);
      gg = WMMA_F16(xAt, loadB(sWih1, 4 + hh), gg);
      gg = WMMA_F16(h1A, fWhh1[4 + hh], gg);
      v8f go = bcast8(b1s[6 + hh]);
      go = WMMA_F16(xAt, loadB(sWih1, 6 + hh), go);
      go = WMMA_F16(h1A, fWhh1[6 + hh], go);

      v8f* cacc = hh ? &c1t1 : &c1t0;
#pragma unroll
      for (int e = 0; e < 8; ++e) {
        float iv = fsig(gi[e]);
        float fv = fsig(gf[e]);
        float gv = ftanh(gg[e]);
        float ov = fsig(go[e]);
        float c  = fv * (*cacc)[e] + iv * gv;
        (*cacc)[e] = c;
        float hv = ov * ftanh(c);
        int m = lh * 8 + e;                 // D layout: VGPR e -> row m
        if (hh == 0)      h1st[m][ln]      = (_Float16)hv;   // j = ln
        else if (ln < 8)  h1st[m][16 + ln] = (_Float16)hv;   // j = 16+ln
      }
    }
    __builtin_amdgcn_wave_barrier();
    h1A = loadA(h1st);   // feed back as A matrix (K=24, pads zero)

    // ---- layer 2: 4 gate tiles (8 valid cols each) ----
    v8f g2i = bcast8(b2s[0]);
    g2i = WMMA_F16(h1A, loadB(sWih2, 0), g2i);
    g2i = WMMA_F16(h2A, fWhh2[0], g2i);
    v8f g2f = bcast8(b2s[1]);
    g2f = WMMA_F16(h1A, loadB(sWih2, 1), g2f);
    g2f = WMMA_F16(h2A, fWhh2[1], g2f);
    v8f g2g = bcast8(b2s[2]);
    g2g = WMMA_F16(h1A, loadB(sWih2, 2), g2g);
    g2g = WMMA_F16(h2A, fWhh2[2], g2g);
    v8f g2o = bcast8(b2s[3]);
    g2o = WMMA_F16(h1A, loadB(sWih2, 3), g2o);
    g2o = WMMA_F16(h2A, fWhh2[3], g2o);

#pragma unroll
    for (int e = 0; e < 8; ++e) {
      float iv = fsig(g2i[e]);
      float fv = fsig(g2f[e]);
      float gv = ftanh(g2g[e]);
      float ov = fsig(g2o[e]);
      float c  = fv * c2t[e] + iv * gv;
      c2t[e] = c;
      float hv = ov * ftanh(c);
      h2last[e] = hv;
      if (ln < 8) h2st[lh * 8 + e][ln] = (_Float16)hv;  // j = ln (<8)
    }
    __builtin_amdgcn_wave_barrier();
    h2A = loadA(h2st);   // recurrent input for next step (K=8, pads zero)
  }

  // ---- final output: out[row][n] = h2(T-1); lane ln<8 holds column n=ln ----
  if (ln < 8) {
#pragma unroll
    for (int e = 0; e < 8; ++e) {
      int row = b0 + lh * 8 + e;
      if (row < batch) out[(size_t)row * HID2 + ln] = h2last[e];
    }
  }
}

extern "C" void kernel_launch(void* const* d_in, const int* in_sizes, int n_in,
                              void* d_out, int out_size, void* d_ws, size_t ws_size,
                              hipStream_t stream) {
  const float* x     = (const float*)d_in[0];
  const float* w_ih1 = (const float*)d_in[1];
  const float* w_hh1 = (const float*)d_in[2];
  const float* b_ih1 = (const float*)d_in[3];
  const float* b_hh1 = (const float*)d_in[4];
  const float* w_ih2 = (const float*)d_in[5];
  const float* w_hh2 = (const float*)d_in[6];
  const float* b_ih2 = (const float*)d_in[7];
  const float* b_hh2 = (const float*)d_in[8];
  float* out = (float*)d_out;

  int batch  = in_sizes[0] / ROWSTRIDE;          // 262144
  int blocks = (batch + 127) / 128;              // 128 rows per 256-thread block

  hipLaunchKernelGGL(lstm2_wmma_kernel, dim3(blocks), dim3(256), 0, stream,
                     x, w_ih1, w_hh1, b_ih1, b_hh1,
                     w_ih2, w_hh2, b_ih2, b_hh2, out, batch);
}
